// GNNProcessor_37288906064620
// MI455X (gfx1250) — compile-verified, matrix operand order checked by reading, not run on hardware
//
#include <hip/hip_runtime.h>

// ---------------------------------------------------------------------------
// GNN processor for MI455X (gfx1250): bf16 WMMA GEMMs, fp32 accumulate/LN.
// ---------------------------------------------------------------------------

typedef __attribute__((ext_vector_type(16))) __bf16 v16bf;
typedef __attribute__((ext_vector_type(8)))  float  v8f;

#define N_NODES 40000
#define N_EDGES 320000
#define HID     128
#define HSTRIDE 136   // bf16 elements, 272 B = 16*17 -> keeps b128 alignment
#define OSTRIDE 132   // fp32 elements (scalar access only)

__device__ __forceinline__ float silu_f(float v) {
    // v * sigmoid(v); v_exp_f32 + v_rcp_f32 instead of IEEE-div sequence
    return v * __builtin_amdgcn_rcpf(1.0f + __expf(-v));
}

// A fragment: 16x32 bf16 tile. lane: m = lane&15, kbase = (lane>>4)*8.
// elements 0..7 at k = kbase..kbase+7 ; elements 8..15 at k = kbase+16..kbase+23
__device__ __forceinline__ v16bf load_a_frag(const __bf16* p) {
    union { uint4 u[2]; v16bf v; } f;
    f.u[0] = *(const uint4*)(p);
    f.u[1] = *(const uint4*)(p + 16);
    return f.v;
}

// B fragment: pre-packed contiguous 16 bf16 per lane (32 B) -> two b128 loads
__device__ __forceinline__ v16bf load_b_frag(const __bf16* p) {
    union { uint4 u[2]; v16bf v; } f;
    f.u[0] = ((const uint4*)p)[0];
    f.u[1] = ((const uint4*)p)[1];
    return f.v;
}

// One GEMM stage: [64 x K] (LDS bf16) @ [K x 128] (packed bf16) + bias (+SiLU)
// 4 waves, wave w owns rows 16w..16w+15; 8 column tiles of 16.
template<int K, bool ACT, typename OT>
__device__ __forceinline__ void gemm_stage(
    const __bf16* __restrict__ Ash, int astride,
    const __bf16* __restrict__ Wp,
    const float*  __restrict__ bias,
    OT* __restrict__ Hsh, int hstride)
{
    constexpr int Kt = K / 32;
    const int tid  = threadIdx.x;
    const int lane = tid & 31;
    const int wave = tid >> 5;
    const int row0 = wave * 16;
    const int nl   = lane & 15;        // A: row-in-tile ; D: column-in-tile
    const int hb   = (lane >> 4) * 8;  // A: k base      ; D: row base
    const __bf16* arow = Ash + (size_t)(row0 + nl) * astride + hb;

    for (int nt = 0; nt < 8; ++nt) {
        v8f acc = {};
        const __bf16* wptr = Wp + ((size_t)nt * Kt * 32 + lane) * 16;
#pragma unroll
        for (int kt = 0; kt < Kt; ++kt) {
            v16bf a = load_a_frag(arow + kt * 32);
            v16bf b = load_b_frag(wptr + (size_t)kt * 32 * 16);
            acc = __builtin_amdgcn_wmma_f32_16x16x32_bf16(
                false, a, false, b, (short)0, acc, false, false);
        }
        const int   n  = nt * 16 + nl;
        const float bn = bias[n];
#pragma unroll
        for (int v = 0; v < 8; ++v) {
            float val = acc[v] + bn;
            if (ACT) val = silu_f(val);
            Hsh[(size_t)(row0 + hb + v) * hstride + n] = (OT)val;
        }
    }
}

// 3-stage MLP on a 64-row tile; OUT gets pre-LayerNorm fp32 result.
template<int K1>
__device__ __forceinline__ void run_mlp(
    const __bf16* Ash, int astride,
    __bf16* H1, __bf16* H2, float* OUT,
    const __bf16* W1p, const __bf16* W2p, const __bf16* W3p,
    const float* b1, const float* b2, const float* b3)
{
    gemm_stage<K1,  true,  __bf16>(Ash, astride, W1p, b1, H1, HSTRIDE);
    __syncthreads();
    gemm_stage<128, true,  __bf16>(H1, HSTRIDE, W2p, b2, H2, HSTRIDE);
    __syncthreads();
    gemm_stage<128, false, float >(H2, HSTRIDE, W3p, b3, OUT, OSTRIDE);
    __syncthreads();
}

// ---------------------------------------------------------------------------
// LayerNorm epilogue across all 128 threads: 2 threads per row, 64 cols each.
// One-pass sum / sum-of-squares, LDS pair-reduction, fused residual + scatter.
// ---------------------------------------------------------------------------
template<bool HAS_RES, bool HAS_AGG>
__device__ __forceinline__ void ln_epilogue(
    const float* __restrict__ OUT, float* __restrict__ red,
    const float* __restrict__ gam, const float* __restrict__ bet,
    const float* __restrict__ res_base, float* __restrict__ out_base,
    long long rowbase,
    float* __restrict__ agg, const int* __restrict__ dIdx)
{
    const int tid  = threadIdx.x;
    const int r    = tid & 63;
    const int half = tid >> 6;
    const int c0   = half * 64;
    const float* o = OUT + r * OSTRIDE + c0;

    float s = 0.0f, q = 0.0f;
    for (int c = 0; c < 64; ++c) { const float v = o[c]; s += v; q += v * v; }
    red[tid] = s;
    red[128 + tid] = q;
    __syncthreads();

    const float sum = red[r] + red[r + 64];
    const float sq  = red[128 + r] + red[128 + r + 64];
    const float m   = sum * (1.0f / HID);
    const float var = sq * (1.0f / HID) - m * m;
    const float rinv = __builtin_amdgcn_rsqf(var + 1e-5f);

    const long long row = rowbase + r;
    const float* res = HAS_RES ? (res_base + row * HID + c0) : nullptr;
    float* outp = out_base + row * HID + c0;
    float* aggp = HAS_AGG ? (agg + (size_t)dIdx[r] * HID + c0) : nullptr;

    for (int c = 0; c < 64; ++c) {
        float val = (o[c] - m) * rinv * gam[c0 + c] + bet[c0 + c];
        if (HAS_RES) val += res[c];
        outp[c] = val;
        if (HAS_AGG) unsafeAtomicAdd(&aggp[c], val);   // global_atomic_add_f32
    }
}

// ---------------------------------------------------------------------------
// Weight packing: fp32 [K x 128] row-major -> bf16 WMMA B fragments.
// frag = nt*(Kpad/32)+kt ; per lane 16 contiguous bf16. Zero-pads k >= K.
// ---------------------------------------------------------------------------
__global__ __launch_bounds__(128) void pack_w_kernel(
    const float* __restrict__ W, int K, int Kpad, __bf16* __restrict__ out)
{
    const int gid = blockIdx.x * blockDim.x + threadIdx.x;
    const int Kt = Kpad / 32;
    const int total = Kt * 8 * 32;
    if (gid >= total) return;
    const int lane = gid & 31;
    const int frag = gid >> 5;
    const int nt = frag / Kt, kt = frag % Kt;
    const int n  = nt * 16 + (lane & 15);
    const int kb = kt * 32 + ((lane >> 4) * 8);
    __bf16* o = out + ((size_t)frag * 32 + lane) * 16;
#pragma unroll
    for (int e = 0; e < 16; ++e) {
        const int k = kb + (e < 8 ? e : 8 + e); // 8..15 -> kb+16..kb+23
        const float v = (k < K) ? W[(size_t)k * HID + n] : 0.0f;
        o[e] = (__bf16)v;
    }
}

__global__ __launch_bounds__(256) void zero_kernel(float* __restrict__ p, size_t n) {
    size_t i = (size_t)blockIdx.x * blockDim.x + threadIdx.x;
    if (i < n) p[i] = 0.0f;
}

// ---------------------------------------------------------------------------
// Embedding MLP: ea = LN(mlp(edge_attr)), K = 8 padded to 32.
// ---------------------------------------------------------------------------
__global__ __launch_bounds__(128) void emb_kernel(
    const float* __restrict__ edge_attr,
    const __bf16* __restrict__ W1p, const __bf16* __restrict__ W2p,
    const __bf16* __restrict__ W3p,
    const float* __restrict__ b1, const float* __restrict__ b2,
    const float* __restrict__ b3,
    const float* __restrict__ gam, const float* __restrict__ bet,
    float* __restrict__ ea_out)
{
    extern __shared__ char smem[];
    const int ASTR = 32 + 8;
    __bf16* Ash = (__bf16*)smem;
    __bf16* H1  = Ash + 64 * ASTR;
    __bf16* H2  = H1 + 64 * HSTRIDE;
    float*  OUT = (float*)(H2 + 64 * HSTRIDE);
    float*  RED = OUT + 64 * OSTRIDE;

    const int tid = threadIdx.x;
    const long long ebase = (long long)blockIdx.x * 64;

    for (int i = tid; i < 64 * 32; i += 128) {
        const int r = i >> 5, c = i & 31;
        const float v = (c < 8) ? edge_attr[(ebase + r) * 8 + c] : 0.0f;
        Ash[r * ASTR + c] = (__bf16)v;
    }
    __syncthreads();

    run_mlp<32>(Ash, ASTR, H1, H2, OUT, W1p, W2p, W3p, b1, b2, b3);

    ln_epilogue<false, false>(OUT, RED, gam, bet, nullptr, ea_out, ebase,
                              nullptr, nullptr);
}

// ---------------------------------------------------------------------------
// Edge MLP: e' = LN(mlp(cat(x[dst], x[src], ea))) + ea ; scatter-add into agg.
// ---------------------------------------------------------------------------
__global__ __launch_bounds__(128) void edge_kernel(
    const float* __restrict__ x, const float* __restrict__ ea_in,
    const int* __restrict__ src, const int* __restrict__ dst,
    const __bf16* __restrict__ W1p, const __bf16* __restrict__ W2p,
    const __bf16* __restrict__ W3p,
    const float* __restrict__ b1, const float* __restrict__ b2,
    const float* __restrict__ b3,
    const float* __restrict__ gam, const float* __restrict__ bet,
    float* __restrict__ ea_out, float* __restrict__ agg)
{
    extern __shared__ char smem[];
    const int ASTR = 384 + 8;
    __bf16* Ash = (__bf16*)smem;
    __bf16* H1  = Ash + 64 * ASTR;
    __bf16* H2  = H1 + 64 * HSTRIDE;
    float*  OUT = (float*)(H2 + 64 * HSTRIDE);
    float*  RED = OUT + 64 * OSTRIDE;
    __shared__ int sIdx[64], dIdx[64];

    const int tid = threadIdx.x;
    const long long ebase = (long long)blockIdx.x * 64;

    if (tid < 64) {
        sIdx[tid] = src[ebase + tid];
        dIdx[tid] = dst[ebase + tid];
    }
    __syncthreads();

    for (int i = tid; i < 64 * 384; i += 128) {
        const int r = i / 384, c = i - r * 384;
        float v;
        if (c < 128)      v = x[(size_t)dIdx[r] * HID + c];
        else if (c < 256) v = x[(size_t)sIdx[r] * HID + (c - 128)];
        else              v = ea_in[(ebase + r) * HID + (c - 256)];
        Ash[r * ASTR + c] = (__bf16)v;
    }
    __syncthreads();

    run_mlp<384>(Ash, ASTR, H1, H2, OUT, W1p, W2p, W3p, b1, b2, b3);

    ln_epilogue<true, true>(OUT, RED, gam, bet, ea_in, ea_out, ebase,
                            agg, dIdx);
}

// ---------------------------------------------------------------------------
// Node MLP: x' = LN(mlp(cat(x, agg))) + x
// ---------------------------------------------------------------------------
__global__ __launch_bounds__(128) void node_kernel(
    const float* __restrict__ x, const float* __restrict__ agg,
    const __bf16* __restrict__ W1p, const __bf16* __restrict__ W2p,
    const __bf16* __restrict__ W3p,
    const float* __restrict__ b1, const float* __restrict__ b2,
    const float* __restrict__ b3,
    const float* __restrict__ gam, const float* __restrict__ bet,
    float* __restrict__ x_out)
{
    extern __shared__ char smem[];
    const int ASTR = 256 + 8;
    __bf16* Ash = (__bf16*)smem;
    __bf16* H1  = Ash + 64 * ASTR;
    __bf16* H2  = H1 + 64 * HSTRIDE;
    float*  OUT = (float*)(H2 + 64 * HSTRIDE);
    float*  RED = OUT + 64 * OSTRIDE;

    const int tid = threadIdx.x;
    const long long nbase = (long long)blockIdx.x * 64;

    for (int i = tid; i < 64 * 256; i += 128) {
        const int r = i >> 8, c = i & 255;
        const float v = (c < 128) ? x[(nbase + r) * HID + c]
                                  : agg[(nbase + r) * HID + (c - 128)];
        Ash[r * ASTR + c] = (__bf16)v;
    }
    __syncthreads();

    run_mlp<256>(Ash, ASTR, H1, H2, OUT, W1p, W2p, W3p, b1, b2, b3);

    ln_epilogue<true, false>(OUT, RED, gam, bet, x, x_out, nbase,
                             nullptr, nullptr);
}

// ---------------------------------------------------------------------------
extern "C" void kernel_launch(void* const* d_in, const int* in_sizes, int n_in,
                              void* d_out, int out_size, void* d_ws, size_t ws_size,
                              hipStream_t stream) {
    (void)in_sizes; (void)n_in; (void)out_size; (void)ws_size;
    // jax tree-flatten order (dict keys sorted): edge, emb, node; within each
    // MLP: W1, W2, W3, b1, b2, b3, be, g
    const float* x         = (const float*)d_in[0];
    const float* edge_attr = (const float*)d_in[1];
    const int*   eidx      = (const int*)d_in[2];
    const int*   src = eidx;            // edge_index[0]
    const int*   dst = eidx + N_EDGES;  // edge_index[1]

    const float* eW1 = (const float*)d_in[3];
    const float* eW2 = (const float*)d_in[4];
    const float* eW3 = (const float*)d_in[5];
    const float* eb1 = (const float*)d_in[6];
    const float* eb2 = (const float*)d_in[7];
    const float* eb3 = (const float*)d_in[8];
    const float* ebe = (const float*)d_in[9];
    const float* eg  = (const float*)d_in[10];

    const float* mW1 = (const float*)d_in[11];
    const float* mW2 = (const float*)d_in[12];
    const float* mW3 = (const float*)d_in[13];
    const float* mb1 = (const float*)d_in[14];
    const float* mb2 = (const float*)d_in[15];
    const float* mb3 = (const float*)d_in[16];
    const float* mbe = (const float*)d_in[17];
    const float* mg  = (const float*)d_in[18];

    const float* nW1 = (const float*)d_in[19];
    const float* nW2 = (const float*)d_in[20];
    const float* nW3 = (const float*)d_in[21];
    const float* nb1 = (const float*)d_in[22];
    const float* nb2 = (const float*)d_in[23];
    const float* nb3 = (const float*)d_in[24];
    const float* nbe = (const float*)d_in[25];
    const float* ng  = (const float*)d_in[26];

    // ---- workspace carve-up ----
    char* ws = (char*)d_ws;
    size_t off = 0;
    auto carve = [&](size_t bytes) { char* p = ws + off; off += (bytes + 255) & ~(size_t)255; return p; };

    __bf16* pw_m1 = (__bf16*)carve(32  * 128 * 2);
    __bf16* pw_m2 = (__bf16*)carve(128 * 128 * 2);
    __bf16* pw_m3 = (__bf16*)carve(128 * 128 * 2);
    __bf16 *pw_e1[2], *pw_e2[2], *pw_e3[2], *pw_n1[2], *pw_n2[2], *pw_n3[2];
    for (int l = 0; l < 2; ++l) {
        pw_e1[l] = (__bf16*)carve(384 * 128 * 2);
        pw_e2[l] = (__bf16*)carve(128 * 128 * 2);
        pw_e3[l] = (__bf16*)carve(128 * 128 * 2);
    }
    for (int l = 0; l < 2; ++l) {
        pw_n1[l] = (__bf16*)carve(256 * 128 * 2);
        pw_n2[l] = (__bf16*)carve(128 * 128 * 2);
        pw_n3[l] = (__bf16*)carve(128 * 128 * 2);
    }
    float* eaA = (float*)carve((size_t)N_EDGES * HID * 4);
    float* eaB = (float*)carve((size_t)N_EDGES * HID * 4);
    float* agg = (float*)carve((size_t)N_NODES * HID * 4);
    float* x1  = (float*)carve((size_t)N_NODES * HID * 4);

    // ---- LDS budgets (WGP has 320 KB; +1 KB reduction buffer) ----
    const int RED_B    = 256 * 4;
    const int EMB_LDS  = 64 * (32  + 8) * 2 + 2 * 64 * HSTRIDE * 2 + 64 * OSTRIDE * 4 + RED_B;
    const int EDGE_LDS = 64 * (384 + 8) * 2 + 2 * 64 * HSTRIDE * 2 + 64 * OSTRIDE * 4 + RED_B;
    const int NODE_LDS = 64 * (256 + 8) * 2 + 2 * 64 * HSTRIDE * 2 + 64 * OSTRIDE * 4 + RED_B;
    hipFuncSetAttribute((const void*)emb_kernel,  hipFuncAttributeMaxDynamicSharedMemorySize, EMB_LDS);
    hipFuncSetAttribute((const void*)edge_kernel, hipFuncAttributeMaxDynamicSharedMemorySize, EDGE_LDS);
    hipFuncSetAttribute((const void*)node_kernel, hipFuncAttributeMaxDynamicSharedMemorySize, NODE_LDS);

    // ---- pack weights (fp32 -> bf16 fragment order) ----
    auto pack = [&](const float* W, int K, int Kpad, __bf16* out) {
        const int total = (Kpad / 32) * 8 * 32;
        pack_w_kernel<<<(total + 127) / 128, 128, 0, stream>>>(W, K, Kpad, out);
    };
    pack(mW1, 8,   32,  pw_m1);
    pack(mW2, 128, 128, pw_m2);
    pack(mW3, 128, 128, pw_m3);
    for (int l = 0; l < 2; ++l) {
        pack(eW1 + (size_t)l * 384 * 128, 384, 384, pw_e1[l]);
        pack(eW2 + (size_t)l * 128 * 128, 128, 128, pw_e2[l]);
        pack(eW3 + (size_t)l * 128 * 128, 128, 128, pw_e3[l]);
        pack(nW1 + (size_t)l * 256 * 128, 256, 256, pw_n1[l]);
        pack(nW2 + (size_t)l * 128 * 128, 128, 128, pw_n2[l]);
        pack(nW3 + (size_t)l * 128 * 128, 128, 128, pw_n3[l]);
    }

    // ---- ea = emb MLP(edge_attr) ----
    emb_kernel<<<N_EDGES / 64, 128, EMB_LDS, stream>>>(
        edge_attr, pw_m1, pw_m2, pw_m3, mb1, mb2, mb3, mg, mbe, eaA);

    // ---- message-passing layers ----
    const size_t aggN = (size_t)N_NODES * HID;
    for (int l = 0; l < 2; ++l) {
        const float* xin  = (l == 0) ? x   : x1;
        float*       xout = (l == 0) ? x1  : (float*)d_out;
        const float* ein  = (l == 0) ? eaA : eaB;
        float*       eout = (l == 0) ? eaB : eaA;

        zero_kernel<<<(int)((aggN + 255) / 256), 256, 0, stream>>>(agg, aggN);

        edge_kernel<<<N_EDGES / 64, 128, EDGE_LDS, stream>>>(
            xin, ein, src, dst,
            pw_e1[l], pw_e2[l], pw_e3[l],
            eb1 + l * HID, eb2 + l * HID, eb3 + l * HID,
            eg + l * HID, ebe + l * HID,
            eout, agg);

        node_kernel<<<N_NODES / 64, 128, NODE_LDS, stream>>>(
            xin, agg,
            pw_n1[l], pw_n2[l], pw_n3[l],
            nb1 + l * HID, nb2 + l * HID, nb3 + l * HID,
            ng + l * HID, nbe + l * HID,
            xout);
    }
}
